// PGC_68315749810482
// MI455X (gfx1250) — compile-verified, matrix operand order checked by reading, not run on hardware
//
#include <hip/hip_runtime.h>
#include <hip/hip_bf16.h>
#include <math.h>

// ---------------- problem constants (match reference) ----------------
#define LDF       128          // fixed row stride for all feature buffers
#define N_NODES   50000        // multiple of 16 (3125 tiles)
#define N_EDGES   800000
#define BATCHSZ   4
#define NCX       19
#define NCY       20
#define NC5       380          // NCX*NCY
#define NP_POOL   (BATCHSZ*NC5) // 1520, multiple of 16 (95 tiles)
#define NC7       16
#define V7X       76.0f
#define V7Y       60.0f
#define ENC_NEG_INF 0x007FFFFFu

typedef __attribute__((ext_vector_type(16))) _Float16 v16h;
typedef __attribute__((ext_vector_type(8)))  _Float16 v8h;
typedef __attribute__((ext_vector_type(8)))  float    v8f;

// Monotone order-preserving float<->u32 encoding for atomic max.
__device__ __forceinline__ unsigned encf(float f) {
  unsigned u = __float_as_uint(f);
  return (u & 0x80000000u) ? ~u : (u | 0x80000000u);
}
__device__ __forceinline__ float decf(unsigned e) {
  return __uint_as_float((e & 0x80000000u) ? (e & 0x7FFFFFFFu) : ~e);
}

// ---------------- f16 staging (removes all guards from the WMMA loop) ----------------
// X (f32, row stride ldx, fin valid cols) -> X16 (f16, row stride LDF, zero-padded)
__global__ void prep_x(const float* __restrict__ X, _Float16* __restrict__ X16,
                       int nrows, int fin, int ldx) {
  int t = blockIdx.x * blockDim.x + threadIdx.x;
  if (t >= nrows * LDF) return;
  int r = t / LDF, c = t - r * LDF;
  X16[t] = (c < fin) ? (_Float16)X[(long long)r * ldx + c] : (_Float16)0.0f;
}
// W (f32, [K,Nout]) -> Wt (f16, [Nout,Kpad]) transposed + K zero-padded
__global__ void prep_w(const float* __restrict__ W, _Float16* __restrict__ Wt,
                       int K, int Kpad, int Nout) {
  int t = blockIdx.x * blockDim.x + threadIdx.x;
  if (t >= Nout * Kpad) return;
  int n = t / Kpad, k = t - n * Kpad;
  Wt[t] = (k < K) ? (_Float16)W[(long long)k * Nout + n] : (_Float16)0.0f;
}

// ---------------- WMMA GEMM: OUT[M,Nout] = X16[M,Kpad] * Wt^T + bias ----------------
// One wave computes one 16x16 tile via v_wmma_f32_16x16x32_f16.
// A fragment (16-bit A 16x32): lane l (l16=l&15 row, half=l>>4):
//   a[0..7]  = A[m][half*8   .. half*8+7]
//   a[8..15] = A[m][16+half*8 .. 16+half*8+7]
// B fragment (32x16): b[e] = B[kmap(e)][n=l16]; with Wt row-major [Nout,Kpad]
// both fragments are two contiguous 16-byte loads -> global_load_b128, no guards.
// C/D: acc[r] = D[m0 + r + 8*half][n0 + l16]
__global__ void gemm_wmma(const _Float16* __restrict__ X16, const _Float16* __restrict__ Wt,
                          const float* __restrict__ bias, float* __restrict__ OUT,
                          int M, int Kpad, int Nout) {
  const int lane = threadIdx.x & 31;
  const int wave = blockIdx.x * (blockDim.x >> 5) + (threadIdx.x >> 5);
  const int ntiles = Nout >> 4;
  const int mt = wave / ntiles;
  const int nt = wave - mt * ntiles;
  const int m0 = mt << 4, n0 = nt << 4;
  if (m0 >= M) return;                 // uniform per-wave exit
  const int half = lane >> 4;
  const int l16  = lane & 15;
  const _Float16* arow = X16 + (long long)(m0 + l16) * LDF + half * 8;
  const _Float16* brow = Wt + (long long)(n0 + l16) * Kpad + half * 8;
  v8f acc = {};
  for (int k0 = 0; k0 < Kpad; k0 += 32) {
    const v8h a0 = *(const v8h*)(arow + k0);
    const v8h a1 = *(const v8h*)(arow + k0 + 16);
    const v8h b0 = *(const v8h*)(brow + k0);
    const v8h b1 = *(const v8h*)(brow + k0 + 16);
    v16h a = __builtin_shufflevector(a0, a1, 0, 1, 2, 3, 4, 5, 6, 7,
                                     8, 9, 10, 11, 12, 13, 14, 15);
    v16h b = __builtin_shufflevector(b0, b1, 0, 1, 2, 3, 4, 5, 6, 7,
                                     8, 9, 10, 11, 12, 13, 14, 15);
    acc = __builtin_amdgcn_wmma_f32_16x16x32_f16(
        /*neg_a=*/false, a, /*neg_b=*/false, b,
        /*c_mod=*/(short)0, acc, /*reuse_a=*/false, /*reuse_b=*/false);
  }
  const float bb = bias[n0 + l16];
#pragma unroll
  for (int r = 0; r < 8; ++r)
    OUT[(long long)(m0 + r + 8 * half) * LDF + (n0 + l16)] = acc[r] + bb;
}

// ---------------- elementwise / util kernels ----------------
__global__ void fill_u32(unsigned* __restrict__ p, unsigned v, int n) {
  int t = blockIdx.x * blockDim.x + threadIdx.x;
  if (t < n) p[t] = v;
}
__global__ void copy_f32(float* __restrict__ d, const float* __restrict__ s, int n) {
  int t = blockIdx.x * blockDim.x + threadIdx.x;
  if (t < n) d[t] = s[t];
}
__global__ void add_f32(float* __restrict__ d, const float* __restrict__ s, int n) {
  int t = blockIdx.x * blockDim.x + threadIdx.x;
  if (t < n) d[t] += s[t];
}

// ---------------- edge scatter: agg[d] = max(agg[d], mf[s] + delta[d]) ----------------
__global__ void edge_scatter_max(const int* __restrict__ src, const int* __restrict__ dst,
                                 int nedges, int groups,
                                 const float* __restrict__ mf, const float* __restrict__ delta,
                                 unsigned* __restrict__ agg) {
  int t = blockIdx.x * blockDim.x + threadIdx.x;
  int total = nedges * groups;
  if (t >= total) return;
  int e = t / groups, g = t - e * groups;
  int s = src[e], d = dst[e];
  if (s == d) return;                  // drop self-loops (reference: -inf mask)
  const float4 mv = *(const float4*)(mf + (long long)s * LDF + g * 4);
  const float4 dv = *(const float4*)(delta + (long long)d * LDF + g * 4);
  unsigned* ap = agg + (long long)d * LDF + g * 4;
  atomicMax(ap + 0, encf(mv.x + dv.x));
  atomicMax(ap + 1, encf(mv.y + dv.y));
  atomicMax(ap + 2, encf(mv.z + dv.z));
  atomicMax(ap + 3, encf(mv.w + dv.w));
}

// ---------------- decode agg, fix isolated nodes, add root transform, ELU ----------------
__global__ void combine_elu(const unsigned* __restrict__ agg, const float* __restrict__ xg,
                            float* __restrict__ h, int n, int F) {
  int t = blockIdx.x * blockDim.x + threadIdx.x;
  if (t >= n * F) return;
  int r = t / F, c = t - r * F;
  float a = decf(agg[(long long)r * LDF + c]);
  if (!__builtin_isfinite(a)) a = 0.0f;   // isolated node -> 0
  float v = a + xg[(long long)r * LDF + c];
  h[(long long)r * LDF + c] = (v > 0.0f) ? v : expm1f(v);  // ELU
}

// ---------------- batchnorm stats: one block per channel, LDS tree reduce ----------------
__global__ void bn_stats(const float* __restrict__ h, const float* __restrict__ mask,
                         int n, float* __restrict__ mu, float* __restrict__ var) {
  __shared__ float ss[256], s2[256], sc[256];
  const int c = blockIdx.x;
  float a = 0.f, b = 0.f, m = 0.f;
  for (int r = threadIdx.x; r < n; r += blockDim.x) {
    float w = mask ? mask[r] : 1.0f;
    float v = h[(long long)r * LDF + c];
    a += v * w; b += v * v * w; m += w;
  }
  ss[threadIdx.x] = a; s2[threadIdx.x] = b; sc[threadIdx.x] = m;
  __syncthreads();
  for (int o = 128; o > 0; o >>= 1) {
    if ((int)threadIdx.x < o) {
      ss[threadIdx.x] += ss[threadIdx.x + o];
      s2[threadIdx.x] += s2[threadIdx.x + o];
      sc[threadIdx.x] += sc[threadIdx.x + o];
    }
    __syncthreads();
  }
  if (threadIdx.x == 0) {
    float cnt = fmaxf(sc[0], 1.0f);
    float mm = ss[0] / cnt;
    mu[c] = mm;
    var[c] = fmaxf(s2[0] / cnt - mm * mm, 0.0f);
  }
}

__global__ void bn_apply(float* __restrict__ h, const float* __restrict__ mu,
                         const float* __restrict__ var, const float* __restrict__ g,
                         const float* __restrict__ b, const float* __restrict__ mask,
                         int n, int F) {
  int t = blockIdx.x * blockDim.x + threadIdx.x;
  if (t >= n * F) return;
  int r = t / F, c = t - r * F;
  float v = h[(long long)r * LDF + c];
  v = (v - mu[c]) * rsqrtf(var[c] + 1e-5f) * g[c] + b[c];
  if (mask) v *= mask[r];
  h[(long long)r * LDF + c] = v;
}

// ---------------- voxel pooling ----------------
__global__ void compute_clus(const float* __restrict__ pos, const int* __restrict__ batch,
                             int* __restrict__ clus, int n) {
  int i = blockIdx.x * blockDim.x + threadIdx.x;
  if (i >= n) return;
  int cx = min(max((int)(pos[i * 3 + 0] / 16.0f), 0), NCX - 1);
  int cy = min(max((int)(pos[i * 3 + 1] / 12.0f), 0), NCY - 1);
  clus[i] = batch[i] * NC5 + cy * NCX + cx;
}

__global__ void pool_pos(const float* __restrict__ pos, const int* __restrict__ clus,
                         float* __restrict__ cnt, float* __restrict__ psum, int n) {
  int i = blockIdx.x * blockDim.x + threadIdx.x;
  if (i >= n) return;
  int c = clus[i];
  atomicAdd(cnt + c, 1.0f);
  atomicAdd(psum + c * 3 + 0, pos[i * 3 + 0]);
  atomicAdd(psum + c * 3 + 1, pos[i * 3 + 1]);
  atomicAdd(psum + c * 3 + 2, pos[i * 3 + 2]);
}

__global__ void pool_feat(const float* __restrict__ h, const int* __restrict__ clus,
                          unsigned* __restrict__ hpenc, int n, int groups) {
  int t = blockIdx.x * blockDim.x + threadIdx.x;
  if (t >= n * groups) return;
  int i = t / groups, g = t - i * groups;
  const float4 v = *(const float4*)(h + (long long)i * LDF + g * 4);
  unsigned* ap = hpenc + (long long)clus[i] * LDF + g * 4;
  atomicMax(ap + 0, encf(v.x));
  atomicMax(ap + 1, encf(v.y));
  atomicMax(ap + 2, encf(v.z));
  atomicMax(ap + 3, encf(v.w));
}

__global__ void pool_fixup(const unsigned* __restrict__ hpenc, const float* __restrict__ cnt,
                           const float* __restrict__ psum, float* __restrict__ hp,
                           float* __restrict__ occ, float* __restrict__ posp, int np, int F) {
  int t = blockIdx.x * blockDim.x + threadIdx.x;
  if (t >= np * F) return;
  int r = t / F, c = t - r * F;
  float o = (cnt[r] > 0.0f) ? 1.0f : 0.0f;
  float v = decf(hpenc[(long long)r * LDF + c]);
  if (o == 0.0f || !__builtin_isfinite(v)) v = 0.0f;
  hp[(long long)r * LDF + c] = v;
  if (c == 0) {
    occ[r] = o;
    float cc = fmaxf(cnt[r], 1.0f);
    posp[r * 3 + 0] = psum[r * 3 + 0] / cc;
    posp[r * 3 + 1] = psum[r * 3 + 1] / cc;
    posp[r * 3 + 2] = psum[r * 3 + 2] / cc;
  }
}

__global__ void edge_remap(const int* __restrict__ src, const int* __restrict__ dst,
                           const int* __restrict__ clus, int* __restrict__ es,
                           int* __restrict__ ed, int ne) {
  int e = blockIdx.x * blockDim.x + threadIdx.x;
  if (e >= ne) return;
  es[e] = clus[src[e]];
  ed[e] = clus[dst[e]];
}

// ---------------- MaxPoolingX: 4x4 grid per sample -> 16 slots ----------------
__global__ void poolx_scatter(const float* __restrict__ h, const float* __restrict__ posp,
                              const float* __restrict__ occ, unsigned* __restrict__ x7enc,
                              int np, int groups) {
  int t = blockIdx.x * blockDim.x + threadIdx.x;
  if (t >= np * groups) return;
  int p = t / groups, g = t - p * groups;
  if (!(occ[p] > 0.0f)) return;
  int c7 = min(max((int)(posp[p * 3 + 1] / V7Y), 0), 3) * 4 +
           min(max((int)(posp[p * 3 + 0] / V7X), 0), 3);
  int seg = (p / NC5) * NC7 + c7;
  const float4 v = *(const float4*)(h + (long long)p * LDF + g * 4);
  unsigned* ap = x7enc + (long long)seg * LDF + g * 4;
  atomicMax(ap + 0, encf(v.x));
  atomicMax(ap + 1, encf(v.y));
  atomicMax(ap + 2, encf(v.z));
  atomicMax(ap + 3, encf(v.w));
}

__global__ void x7_fixup(const unsigned* __restrict__ x7enc, float* __restrict__ x7, int n) {
  int t = blockIdx.x * blockDim.x + threadIdx.x;
  if (t >= n) return;
  float v = decf(x7enc[t]);
  x7[t] = __builtin_isfinite(v) ? v : 0.0f;
}

// ---------------- final FC: [B, 2048] @ [2048, 2] ----------------
__global__ void fc_kernel(const float* __restrict__ x7, const float* __restrict__ Wfc,
                          float* __restrict__ out) {
  __shared__ float red[256];
  int b = blockIdx.x >> 1, o = blockIdx.x & 1;
  float s = 0.0f;
  for (int i = threadIdx.x; i < NC7 * LDF; i += blockDim.x)
    s += x7[b * NC7 * LDF + i] * Wfc[i * 2 + o];
  red[threadIdx.x] = s;
  __syncthreads();
  for (int off = 128; off > 0; off >>= 1) {
    if ((int)threadIdx.x < off) red[threadIdx.x] += red[threadIdx.x + off];
    __syncthreads();
  }
  if (threadIdx.x == 0) out[b * 2 + o] = red[0];
}

// ---------------- driver ----------------
extern "C" void kernel_launch(void* const* d_in, const int* in_sizes, int n_in,
                              void* d_out, int out_size, void* d_ws, size_t ws_size,
                              hipStream_t stream) {
  (void)in_sizes; (void)n_in; (void)out_size; (void)ws_size;
  // Input order = setup_inputs() dict insertion order:
  //   x, pos, [params: per layer Wh,bh,Wf,bf,Wg,bg,bn_g,bn_b x7, then Wfc], edge_index, batch
  const float* x   = (const float*)d_in[0];
  const float* pos = (const float*)d_in[1];
  const float* P[57];
  for (int i = 0; i < 57; ++i) P[i] = (const float*)d_in[2 + i];
  const float* Wfc = P[56];
  const int* edge_index = (const int*)d_in[59];
  const int* batch      = (const int*)d_in[60];
  const int* src = edge_index;
  const int* dst = edge_index + N_EDGES;
  float* out = (float*)d_out;

  // workspace carve (256B aligned chunks)
  char* wsp = (char*)d_ws;
  size_t off = 0;
  auto carve = [&](size_t bytes) -> char* {
    char* p = wsp + off;
    off += (bytes + 255) & ~(size_t)255;
    return p;
  };
  const size_t NF = (size_t)N_NODES * LDF * sizeof(float);
  float*     hA    = (float*)carve(NF);
  float*     hB    = (float*)carve(NF);
  float*     scb   = (float*)carve(NF);
  float*     dlt   = (float*)carve(NF);
  float*     mfb   = (float*)carve(NF);
  float*     xgb   = (float*)carve(NF);
  unsigned*  agg   = (unsigned*)carve(NF);
  _Float16*  x16   = (_Float16*)carve((size_t)N_NODES * LDF * 2);
  _Float16*  wt16  = (_Float16*)carve((size_t)LDF * LDF * 2);
  int*       clus  = (int*)carve((size_t)N_NODES * 4);
  int*       es    = (int*)carve((size_t)N_EDGES * 4);
  int*       ed    = (int*)carve((size_t)N_EDGES * 4);
  float*     cnt   = (float*)carve((size_t)NP_POOL * 4);
  float*     psum  = (float*)carve((size_t)NP_POOL * 12);
  float*     posp  = (float*)carve((size_t)NP_POOL * 12);
  float*     occ   = (float*)carve((size_t)NP_POOL * 4);
  unsigned*  hpenc = (unsigned*)carve((size_t)NP_POOL * LDF * 4);
  float*     hpA   = (float*)carve((size_t)NP_POOL * LDF * 4);
  float*     hpB   = (float*)carve((size_t)NP_POOL * LDF * 4);
  float*     scP   = (float*)carve((size_t)NP_POOL * LDF * 4);
  unsigned*  x7enc = (unsigned*)carve((size_t)BATCHSZ * NC7 * LDF * 4);
  float*     x7    = (float*)carve((size_t)BATCHSZ * NC7 * LDF * 4);
  float*     mu    = (float*)carve(LDF * 4);
  float*     var   = (float*)carve(LDF * 4);

  auto gblk = [](int n) { return (n + 255) / 256; };

  // One PointGNN conv + ELU + (masked) BN.
  auto conv = [&](const float* xin, int ldx, int fin, int fout, int nrows,
                  const int* s, const int* d, int li, const float* mask, float* hout) {
    const float* Wh = P[li * 8 + 0]; const float* bh = P[li * 8 + 1];
    const float* Wf = P[li * 8 + 2]; const float* bf = P[li * 8 + 3];
    const float* Wg = P[li * 8 + 4]; const float* bg = P[li * 8 + 5];
    const float* gg = P[li * 8 + 6]; const float* gb = P[li * 8 + 7];
    const int Kpad = (fin + 31) & ~31;
    // stage activations once (shared by 3 GEMMs)
    prep_x<<<gblk(nrows * LDF), 256, 0, stream>>>(xin, x16, nrows, fin, ldx);
    dim3 gg8(((nrows / 16) * (fout / 16) + 7) / 8);
    const int wtn = fout * Kpad;
    prep_w<<<gblk(wtn), 256, 0, stream>>>(Wh, wt16, fin, Kpad, fout);
    gemm_wmma<<<gg8, 256, 0, stream>>>(x16, wt16, bh, dlt, nrows, Kpad, fout);
    prep_w<<<gblk(wtn), 256, 0, stream>>>(Wf, wt16, fin, Kpad, fout);
    gemm_wmma<<<gg8, 256, 0, stream>>>(x16, wt16, bf, mfb, nrows, Kpad, fout);
    prep_w<<<gblk(wtn), 256, 0, stream>>>(Wg, wt16, fin, Kpad, fout);
    gemm_wmma<<<gg8, 256, 0, stream>>>(x16, wt16, bg, xgb, nrows, Kpad, fout);
    int nfill = nrows * LDF;
    fill_u32<<<gblk(nfill), 256, 0, stream>>>(agg, ENC_NEG_INF, nfill);
    int groups = fout / 4;
    int etot = N_EDGES * groups;
    edge_scatter_max<<<gblk(etot), 256, 0, stream>>>(s, d, N_EDGES, groups, mfb, dlt, agg);
    int tc = nrows * fout;
    combine_elu<<<gblk(tc), 256, 0, stream>>>(agg, xgb, hout, nrows, fout);
    bn_stats<<<fout, 256, 0, stream>>>(hout, mask, nrows, mu, var);
    bn_apply<<<gblk(tc), 256, 0, stream>>>(hout, mu, var, gg, gb, mask, nrows, fout);
  };

  const int nfull = N_NODES * LDF;
  // layers 1..5 on the full graph
  conv(x,  1,   1,   16, N_NODES, src, dst, 0, nullptr, hA);
  conv(hA, LDF, 16,  64, N_NODES, src, dst, 1, nullptr, hB);
  copy_f32<<<gblk(nfull), 256, 0, stream>>>(scb, hB, nfull);      // sc = h2
  conv(hB, LDF, 64,  64, N_NODES, src, dst, 2, nullptr, hA);
  conv(hA, LDF, 64,  64, N_NODES, src, dst, 3, nullptr, hB);
  add_f32<<<gblk(nfull), 256, 0, stream>>>(hB, scb, nfull);       // h4 += sc
  conv(hB, LDF, 64, 128, N_NODES, src, dst, 4, nullptr, hA);

  // voxel pooling (16x12 grid per sample)
  compute_clus<<<gblk(N_NODES), 256, 0, stream>>>(pos, batch, clus, N_NODES);
  fill_u32<<<gblk(NP_POOL), 256, 0, stream>>>((unsigned*)cnt, 0u, NP_POOL);
  fill_u32<<<gblk(NP_POOL * 3), 256, 0, stream>>>((unsigned*)psum, 0u, NP_POOL * 3);
  fill_u32<<<gblk(NP_POOL * LDF), 256, 0, stream>>>(hpenc, ENC_NEG_INF, NP_POOL * LDF);
  pool_pos<<<gblk(N_NODES), 256, 0, stream>>>(pos, clus, cnt, psum, N_NODES);
  pool_feat<<<gblk(N_NODES * 32), 256, 0, stream>>>(hA, clus, hpenc, N_NODES, 32);
  pool_fixup<<<gblk(NP_POOL * LDF), 256, 0, stream>>>(hpenc, cnt, psum, hpA, occ, posp,
                                                      NP_POOL, LDF);
  edge_remap<<<gblk(N_EDGES), 256, 0, stream>>>(src, dst, clus, es, ed, N_EDGES);

  // layers 6..7 on pooled graph (masked BN), residual
  const int npf = NP_POOL * LDF;
  copy_f32<<<gblk(npf), 256, 0, stream>>>(scP, hpA, npf);         // sc = hp
  conv(hpA, LDF, 128, 128, NP_POOL, es, ed, 5, occ, hpB);
  conv(hpB, LDF, 128, 128, NP_POOL, es, ed, 6, occ, hpA);
  add_f32<<<gblk(npf), 256, 0, stream>>>(hpA, scP, npf);          // h7 += sc

  // MaxPoolingX -> [B*16, 128] -> FC -> [B, 2]
  const int x7n = BATCHSZ * NC7 * LDF;
  fill_u32<<<gblk(x7n), 256, 0, stream>>>(x7enc, ENC_NEG_INF, x7n);
  poolx_scatter<<<gblk(NP_POOL * 32), 256, 0, stream>>>(hpA, posp, occ, x7enc, NP_POOL, 32);
  x7_fixup<<<gblk(x7n), 256, 0, stream>>>(x7enc, x7, x7n);
  fc_kernel<<<BATCHSZ * 2, 256, 0, stream>>>(x7, Wfc, out);
}
